// BipartitePooling_51170240365321
// MI455X (gfx1250) — matrix operand order repaired and checked by reading, not compiled, not run on hardware
//
#include <hip/hip_runtime.h>
#include <hip/hip_bf16.h>

// ---------------- fixed problem dims ----------------
#define NB    8
#define NPER  2048
#define NTOT  16384
#define FDIM  128
#define RATIO 32
#define HH    4
#define CC    32
#define TT    256
#define NEG   0.2f

// ---------------- workspace layout (bytes) ----------------
#define OFF_XB  0ul         // x bf16                         : 4 MiB
#define OFF_WL  4194304ul   // W_l bf16, WMMA-B pre-swizzled  : 32 KiB
#define OFF_WR  4227072ul   // W_r bf16, WMMA-B pre-swizzled  : 32 KiB
#define OFF_SB  4259840ul   // seed bf16                      : 8 KiB
#define OFF_HS  4268032ul   // h_src f32                      : 8 MiB
#define OFF_HD  12656640ul  // h_dst f32                      : 16 KiB
#define OFF_LG  12673024ul  // logits f32 [B][H][32][2048]    : 8 MiB
#define OFF_MX  21061632ul  // per-(t,h) max (ordered uint)   : 4 KiB
#define OFF_SM  21065728ul  // per-(t,h) exp-sum f32          : 4 KiB

typedef __attribute__((ext_vector_type(16))) __bf16 v16bf;
typedef __attribute__((ext_vector_type(8)))  __bf16 v8bf;
typedef __attribute__((ext_vector_type(8)))  float  v8f;
typedef __attribute__((ext_vector_type(2)))  float  v2f;

// monotonic unsigned encoding of float for atomicMax (init value 0u < every finite)
__device__ __forceinline__ unsigned fenc(float f) {
    unsigned u = __float_as_uint(f);
    return (u & 0x80000000u) ? ~u : (u | 0x80000000u);
}
__device__ __forceinline__ float fdec(unsigned e) {
    return (e & 0x80000000u) ? __uint_as_float(e ^ 0x80000000u) : __uint_as_float(~e);
}

// ---------------- kernel 1: bf16 staging (+ WMMA-B weight swizzle), init, new_batch ----
// Weight element W[k][n] goes to a lane/halfword slot of the 16x16x32 bf16 WMMA
// B-operand, in contiguous k order (matching the GEMM's A):  wp[((nt*4+kk)*32 + lane)*16 + j],
//   kk=k>>5, hi=(k>>4)&1, j=k&15, nt=n>>4, lane=hi*16+(n&15)
// so the GEMM reads each lane's B fragment as two contiguous b128 loads.
__global__ void convert_init_kernel(const float* __restrict__ x,
                                    const float* __restrict__ seed,
                                    const float* __restrict__ Wl,
                                    const float* __restrict__ Wr,
                                    __bf16* __restrict__ xb, __bf16* __restrict__ sb,
                                    __bf16* __restrict__ wlp, __bf16* __restrict__ wrp,
                                    unsigned* __restrict__ mx, int* __restrict__ out_batch) {
    int i = blockIdx.x * blockDim.x + threadIdx.x;
    if (i < NTOT * FDIM) xb[i] = (__bf16)x[i];
    if (i < FDIM * FDIM) {
        int k = i >> 7, n = i & 127;
        int kk = k >> 5, hi = (k >> 4) & 1, j = k & 15;
        int nt = n >> 4, lane = hi * 16 + (n & 15);
        int pidx = (((nt << 2) + kk) * 32 + lane) * 16 + j;
        wlp[pidx] = (__bf16)Wl[i];
        wrp[pidx] = (__bf16)Wr[i];
    }
    if (i < RATIO * FDIM)    sb[i] = (__bf16)seed[i];
    if (i < NB * HH * RATIO) mx[i] = 0u;
    if (i < TT) out_batch[i] = i >> 5;               // new_batch = repeat(arange(B), 32)
}

// ---------------- kernel 2: bf16 WMMA GEMM  C[M,128] = A[M,128] @ W[128,128] -----------
// one 16x16 output tile per wave; 4 x v_wmma_f32_16x16x32_bf16; all operands via b128 loads
__global__ void wmma_gemm_bf16_kernel(const __bf16* __restrict__ A,
                                      const __bf16* __restrict__ Bp,   // pre-swizzled
                                      float* __restrict__ C) {
    int wave = threadIdx.x >> 5;
    int lane = threadIdx.x & 31;
    int tile = blockIdx.x * 4 + wave;
    int mt = tile >> 3, nt = tile & 7;           // 8 N-tiles (128/16)
    int l15 = lane & 15, hi = lane >> 4;

    const __bf16* arow = A + (size_t)(mt * 16 + l15) * FDIM;

    v8f acc = {};
#pragma unroll
    for (int kk = 0; kk < 4; ++kk) {
        // A 16x32 bf16: lane half `hi` holds K = kk*32 + hi*16 + {0..15}
        union { v16bf v; v8bf h[2]; } a;
        int k0 = kk * 32 + hi * 16;
        a.h[0] = *(const v8bf*)(arow + k0);
        a.h[1] = *(const v8bf*)(arow + k0 + 8);
        // B fragment: 16 contiguous bf16 per lane from the swizzled buffer
        union { v16bf v; v8bf h[2]; } b;
        const v8bf* bp = (const v8bf*)(Bp + (size_t)((((nt << 2) + kk) * 32) + lane) * 16);
        b.h[0] = bp[0];
        b.h[1] = bp[1];
        acc = __builtin_amdgcn_wmma_f32_16x16x32_bf16(
            false, a.v, false, b.v, (short)0, acc, false, false);
    }
    // C/D layout: VGPR v -> row M = v + hi*8, col N = l15
#pragma unroll
    for (int v = 0; v < 8; ++v)
        C[(size_t)(mt * 16 + v + hi * 8) * FDIM + nt * 16 + l15] = acc[v];
}

// ---------------- kernel 3: attention logits + running per-(t,h) max -------------------
// grid: (g,h,chunk-of-256-nodes) = 8*4*8 = 256 blocks of 256 threads
__global__ void logits_kernel(const float* __restrict__ hs, const float* __restrict__ hd,
                              const float* __restrict__ att,
                              float* __restrict__ lg, unsigned* __restrict__ mx) {
    __shared__ float shd[RATIO][CC + 1];
    __shared__ float satt[CC];
    int b = blockIdx.x;
    int g = b >> 5, h = (b >> 3) & 3, chunk = b & 7;
    int tid = threadIdx.x;

    if (tid < CC) satt[tid] = att[h * CC + tid];
    for (int i = tid; i < RATIO * CC; i += 256)
        shd[i >> 5][i & 31] = hd[(size_t)(i >> 5) * FDIM + h * CC + (i & 31)];
    __syncthreads();

    int n = chunk * 256 + tid;
    float v[CC];
    const float4* p = (const float4*)(hs + (size_t)(g * NPER + n) * FDIM + h * CC);
#pragma unroll
    for (int i = 0; i < 8; ++i) ((float4*)v)[i] = p[i];

    int lane = tid & 31;
    float* lgbase = lg + (size_t)((g * HH + h) * RATIO) * NPER + n;
    unsigned* mxbase = mx + (g * HH + h) * RATIO;

    for (int r = 0; r < RATIO; ++r) {
        float s = 0.f;
#pragma unroll
        for (int c = 0; c < CC; ++c) {
            float t = shd[r][c] + v[c];
            t = t > 0.f ? t : NEG * t;
            s = fmaf(satt[c], t, s);
        }
        lgbase[(size_t)r * NPER] = s;                    // coalesced over n
        float wm = s;
#pragma unroll
        for (int off = 16; off > 0; off >>= 1)
            wm = fmaxf(wm, __shfl_xor(wm, off, 32));
        if (lane == 0) atomicMax(&mxbase[r], fenc(wm));
    }
}

// ---------------- kernel 4: softmax denominator per (t,h) ------------------------------
__global__ void softmax_sum_kernel(const float* __restrict__ lg,
                                   const unsigned* __restrict__ mx,
                                   float* __restrict__ sums) {
    __shared__ float sw[8];
    int idx = blockIdx.x;                         // 1024 = (g*4+h)*32 + r
    float m = fdec(mx[idx]);
    const float* base = lg + (size_t)idx * NPER;
    float local = 0.f;
    for (int n = threadIdx.x; n < NPER; n += 256) {
        __builtin_prefetch(base + n + 1024, 0, 1);   // global_prefetch_b8 (stream-ahead)
        local += __expf(base[n] - m);
    }
#pragma unroll
    for (int off = 16; off > 0; off >>= 1) local += __shfl_xor(local, off, 32);
    int wave = threadIdx.x >> 5, lane = threadIdx.x & 31;
    if (lane == 0) sw[wave] = local;
    __syncthreads();
    if (threadIdx.x == 0) {
        float t = 0.f;
#pragma unroll
        for (int w = 0; w < 8; ++w) t += sw[w];
        sums[idx] = t;
    }
}

// ---------------- kernel 5: out = alpha^T @ h_src via f32 WMMA + bias -------------------
// grid: 32 blocks (one per (g,h)); 4 waves = the four 16x16 tiles of the 32x32 output
__global__ void aggregate_wmma_kernel(const float* __restrict__ lg,
                                      const unsigned* __restrict__ mx,
                                      const float* __restrict__ sums,
                                      const float* __restrict__ hs,
                                      const float* __restrict__ bias,
                                      float* __restrict__ out) {
    int g = blockIdx.x >> 2, h = blockIdx.x & 3;
    int wave = threadIdx.x >> 5, lane = threadIdx.x & 31;
    int rt = wave >> 1, nt = wave & 1;
    int l15 = lane & 15, hi = lane >> 4;

    int r  = rt * 16 + l15;                       // A row (same for both lane halves)
    int tb = (g * HH + h) * RATIO + r;
    float m    = fdec(mx[tb]);
    float invs = 1.0f / sums[tb];
    const float* lrow = lg + (size_t)tb * NPER;

    int col = h * CC + nt * 16 + l15;             // B column
    const float* hcol = hs + (size_t)g * NPER * FDIM + col;

    v8f acc = {};
    for (int k = 0; k < NPER; k += 4) {
        int ka = k + hi * 2;                      // f32 A 16x4: lane half holds K = hi*2 + {0,1}
        v2f a, b;
        a[0] = __expf(lrow[ka]     - m) * invs;
        a[1] = __expf(lrow[ka + 1] - m) * invs;
        b[0] = hcol[(size_t)ka * FDIM];           // f32 B 4x16: rows K = hi*2 + {0,1}, col = l15
        b[1] = hcol[(size_t)(ka + 1) * FDIM];
        acc = __builtin_amdgcn_wmma_f32_16x16x4_f32(
            false, a, false, b, (short)0, acc, false, false);
    }

    float bv = bias[col];
#pragma unroll
    for (int v = 0; v < 8; ++v) {
        int t = g * RATIO + rt * 16 + v + hi * 8; // D layout: VGPR v -> row v + hi*8
        out[(size_t)t * FDIM + col] = acc[v] + bv;
    }
}

// ---------------- host launch ----------------
extern "C" void kernel_launch(void* const* d_in, const int* in_sizes, int n_in,
                              void* d_out, int out_size, void* d_ws, size_t ws_size,
                              hipStream_t stream) {
    const float* x    = (const float*)d_in[0];
    // d_in[1] = batch (int32) — regular layout (nodes sorted by graph), unused
    const float* seed = (const float*)d_in[2];
    const float* Wl   = (const float*)d_in[3];
    const float* Wr   = (const float*)d_in[4];
    const float* att  = (const float*)d_in[5];
    const float* bias = (const float*)d_in[6];

    char* ws = (char*)d_ws;
    __bf16*   xb   = (__bf16*)(ws + OFF_XB);
    __bf16*   wlp  = (__bf16*)(ws + OFF_WL);
    __bf16*   wrp  = (__bf16*)(ws + OFF_WR);
    __bf16*   sb   = (__bf16*)(ws + OFF_SB);
    float*    hsrc = (float*)(ws + OFF_HS);
    float*    hdst = (float*)(ws + OFF_HD);
    float*    lgts = (float*)(ws + OFF_LG);
    unsigned* mx   = (unsigned*)(ws + OFF_MX);
    float*    sums = (float*)(ws + OFF_SM);

    float* out       = (float*)d_out;
    int*   out_batch = ((int*)d_out) + TT * FDIM;    // tuple part 2, concatenated flat

    convert_init_kernel<<<8192, 256, 0, stream>>>(x, seed, Wl, Wr, xb, sb, wlp, wrp, mx, out_batch);
    wmma_gemm_bf16_kernel<<<2048, 128, 0, stream>>>(xb, wlp, hsrc);   // h_src: 16384x128
    wmma_gemm_bf16_kernel<<<4,    128, 0, stream>>>(sb, wrp, hdst);   // h_dst: 32x128
    logits_kernel<<<256, 256, 0, stream>>>(hsrc, hdst, att, lgts, mx);
    softmax_sum_kernel<<<1024, 256, 0, stream>>>(lgts, mx, sums);
    aggregate_wmma_kernel<<<32, 128, 0, stream>>>(lgts, mx, sums, hsrc, bias, out);
}
